// T_RED_GNN_v2_20993800142933
// MI455X (gfx1250) — matrix operand order repaired and checked by reading, not compiled, not run on hardware
//
#include <hip/hip_runtime.h>
#include <hip/hip_bf16.h>

typedef __attribute__((ext_vector_type(16))) _Float16 v16h;
typedef __attribute__((ext_vector_type(8)))  float    v8f;

#define N_NODES 200000
#define N_ENT   10000
#define BATCH   256
#define REL_VOCAB 402
#define N_EDGES 1000000
#define HID     20

__device__ __forceinline__ float sigmoidf_(float x) {
    return 1.0f / (1.0f + __expf(-x));
}

// One wave (32 lanes) processes a tile of 16 edges per iteration.
// Attention input (16 edges x 60 feats, padded to 64) -> two 16x32 f16 A slabs.
// W1^T (60x30, padded 64x32) -> four 32x16 f16 B tiles (2 K-slabs x 2 N-tiles).
// 4x v_wmma_f32_16x16x32_f16 per tile, then relu + W2 dot + cross-lane reduce,
// sigmoid, and 20-dim atomic scatter of score*(h_src + r_emb) into hout[dst].
// rela_embed (32KB) and the pre-dereferenced per-batch query embeddings (20KB)
// are staged in LDS: only the h_src gathers and hout atomics touch VMEM/L2.
__global__ __launch_bounds__(256) void redgnn_layer_kernel(
    const float* __restrict__ hin,
    float* __restrict__ hout,
    const int* __restrict__ src_idx,
    const int* __restrict__ rel_idx,
    const int* __restrict__ batch_idx,
    const int* __restrict__ dst_idx,
    const int* __restrict__ query_rel,
    const float* __restrict__ rela,   // [402,20]
    const float* __restrict__ W1,     // [30,60]
    const float* __restrict__ W2)     // [1,30]
{
    __shared__ float s_rela[REL_VOCAB * HID];   // 32160 B
    __shared__ float s_qemb[BATCH * HID];       // 20480 B

    // ---- Stage relation table, then pre-dereferenced query embeddings ----
    for (int i = threadIdx.x; i < REL_VOCAB * HID; i += blockDim.x)
        s_rela[i] = rela[i];
    __syncthreads();
    for (int i = threadIdx.x; i < BATCH * HID; i += blockDim.x) {
        const int b = i / HID, d = i - b * HID;
        s_qemb[i] = s_rela[query_rel[b] * HID + d];
    }
    __syncthreads();

    const int lane   = threadIdx.x & 31;
    const int gwave  = (blockIdx.x * blockDim.x + threadIdx.x) >> 5;
    const int nwaves = (gridDim.x * blockDim.x) >> 5;
    const int m      = lane & 15;        // edge-in-tile this lane serves
    const bool hiHalf = lane >= 16;

    // ---- B operands (W1^T), built once; W1 is tiny and cache-resident ----
    // B 32x16 f16 layout: lanes 0-15 hold K=0..7 & 16..23, lanes 16-31 K=8..15 & 24..31; n = lane%16.
    v16h B00{}, B01{}, B10{}, B11{};
    {
        const int n     = lane & 15;
        const int kbase = hiHalf ? 8 : 0;
        const int n1    = 16 + n;
        #pragma unroll
        for (int e = 0; e < 16; ++e) {
            const int k0 = (e < 8) ? (kbase + e) : (16 + kbase + (e - 8));        // slab 0 K (0..31)
            const int k1 = 32 + k0;   // slab 1 K (32..63)
            float w;
            w = (k0 < 60) ? W1[n * 60 + k0] : 0.f;               B00[e] = (_Float16)w;
            w = (k1 < 60) ? W1[n * 60 + k1] : 0.f;               B10[e] = (_Float16)w;
            w = (n1 < 30 && k0 < 60) ? W1[n1 * 60 + k0] : 0.f;   B01[e] = (_Float16)w;
            w = (n1 < 30 && k1 < 60) ? W1[n1 * 60 + k1] : 0.f;   B11[e] = (_Float16)w;
        }
    }
    const int   nloc = lane & 15;
    const float w2a  = W2[nloc];
    const float w2b  = (16 + nloc < 30) ? W2[16 + nloc] : 0.f;

    const int nTiles = N_EDGES / 16;
    for (int tile = gwave; tile < nTiles; tile += nwaves) {
        const int ebase = tile * 16;
        const int se = src_idx[ebase + m];
        const int re = rel_idx[ebase + m];
        const int be = batch_idx[ebase + m];
        const int de = dst_idx[ebase + m];

        // ---- A operands: 16-bit A 16x32 layout. lane<16: K=0..7 & 16..23;
        //      lane>=16: K=8..15 & 24..31 (element e<8 -> ko+e, else 16+ko+e-8).
        v16h A0{}, A1{};
        const int ko = hiHalf ? 8 : 0;
        #pragma unroll
        for (int e = 0; e < 16; ++e) {
            const int kl = (e < 8) ? (ko + e) : (16 + ko + (e - 8));
            // slab 0: feature kg in 0..31 (h_src from VMEM, r_emb from LDS)
            {
                const int kg = kl;
                float fv = (kg < 20) ? hin[se * HID + kg]
                                     : s_rela[re * HID + (kg - 20)];
                A0[e] = (_Float16)fv;
            }
            // slab 1: feature kg in 32..63 (r_emb / q_emb from LDS)
            {
                const int kg = 32 + kl;
                float fv;
                if (kg < 40)      fv = s_rela[re * HID + (kg - 20)];
                else if (kg < 60) fv = s_qemb[be * HID + (kg - 40)];
                else              fv = 0.f;
                A1[e] = (_Float16)fv;
            }
        }

        // ---- 16x60 @ 60x32 via 4 WMMAs: hid[m][n] = attn_in @ W1^T ----
        v8f c0 = {}, c1 = {};
        c0 = __builtin_amdgcn_wmma_f32_16x16x32_f16(false, A0, false, B00, (short)0, c0, false, false);
        c0 = __builtin_amdgcn_wmma_f32_16x16x32_f16(false, A1, false, B10, (short)0, c0, false, false);
        c1 = __builtin_amdgcn_wmma_f32_16x16x32_f16(false, A0, false, B01, (short)0, c1, false, false);
        c1 = __builtin_amdgcn_wmma_f32_16x16x32_f16(false, A1, false, B11, (short)0, c1, false, false);

        // ---- relu + W2 dot: D layout => lane holds column n=lane%16,
        //      rows v (lo half) / v+8 (hi half). Reduce over n across each half.
        float p[8];
        #pragma unroll
        for (int v = 0; v < 8; ++v) {
            const float h0 = c0[v] > 0.f ? c0[v] : 0.f;
            const float h1 = c1[v] > 0.f ? c1[v] : 0.f;
            float t = h0 * w2a + h1 * w2b;
            #pragma unroll
            for (int mask = 1; mask < 16; mask <<= 1)
                t += __shfl_xor(t, mask, 32);
            p[v] = t;   // lanes 0-15: edge v; lanes 16-31: edge v+8
        }
        // broadcast register q = p[lane&7] (constant-index select chain)
        float q = p[0];
        #pragma unroll
        for (int j = 1; j < 8; ++j) q = ((lane & 7) == j) ? p[j] : q;

        // score_pre for edge m lives on lane (m>>3)*16 + (m&7)
        const float spre  = __shfl(q, ((m >> 3) << 4) + (m & 7), 32);
        const float score = sigmoidf_(spre);

        // ---- msg scatter: lane<16 -> dims 0..9, lane>=16 -> dims 10..19 ----
        const int dbase = hiHalf ? 10 : 0;
        #pragma unroll
        for (int dd = 0; dd < 10; ++dd) {
            const int d = dbase + dd;
            const float val = score * (hin[se * HID + d] + s_rela[re * HID + d]);
            atomicAdd(&hout[de * HID + d], val);
        }
    }
}

__global__ __launch_bounds__(256) void redgnn_final_kernel(
    const float* __restrict__ hidden,
    const int* __restrict__ final_batch,
    const int* __restrict__ final_ent,
    const float* __restrict__ Wc,
    const float* __restrict__ bc,
    float* __restrict__ out)
{
    const int n = blockIdx.x * blockDim.x + threadIdx.x;
    if (n >= N_NODES) return;
    float acc = bc[0];
    #pragma unroll
    for (int d = 0; d < HID; ++d) acc += hidden[n * HID + d] * Wc[d];
    out[final_batch[n] * N_ENT + final_ent[n]] = acc;
}

extern "C" void kernel_launch(void* const* d_in, const int* in_sizes, int n_in,
                              void* d_out, int out_size, void* d_ws, size_t ws_size,
                              hipStream_t stream) {
    const int*   query_rel  = (const int*)  d_in[0];
    const int*   src_idx    = (const int*)  d_in[1];   // [3, 1M]
    const int*   rel_idx    = (const int*)  d_in[2];
    const int*   batch_idx  = (const int*)  d_in[3];
    const int*   dst_idx    = (const int*)  d_in[4];
    const int*   final_b    = (const int*)  d_in[5];
    const int*   final_e    = (const int*)  d_in[6];
    const float* rela_embed = (const float*)d_in[7];
    const float* W1         = (const float*)d_in[8];
    const float* W2         = (const float*)d_in[9];
    const float* Wc         = (const float*)d_in[10];
    const float* bc         = (const float*)d_in[11];
    float* out = (float*)d_out;

    const size_t hbytes = (size_t)N_NODES * HID * sizeof(float);
    float* hA = (float*)d_ws;
    float* hB = hA + (size_t)N_NODES * HID;

    hipMemsetAsync(hA, 0, hbytes, stream);
    hipMemsetAsync(hB, 0, hbytes, stream);

    const dim3 egrid(1024), eblock(256);
    // layer 0: in=hA(zeros), out=hB
    redgnn_layer_kernel<<<egrid, eblock, 0, stream>>>(
        hA, hB, src_idx + 0 * N_EDGES, rel_idx + 0 * N_EDGES,
        batch_idx + 0 * N_EDGES, dst_idx + 0 * N_EDGES,
        query_rel, rela_embed, W1, W2);
    hipMemsetAsync(hA, 0, hbytes, stream);
    // layer 1: in=hB, out=hA
    redgnn_layer_kernel<<<egrid, eblock, 0, stream>>>(
        hB, hA, src_idx + 1 * N_EDGES, rel_idx + 1 * N_EDGES,
        batch_idx + 1 * N_EDGES, dst_idx + 1 * N_EDGES,
        query_rel, rela_embed, W1, W2);
    hipMemsetAsync(hB, 0, hbytes, stream);
    // layer 2: in=hA, out=hB
    redgnn_layer_kernel<<<egrid, eblock, 0, stream>>>(
        hA, hB, src_idx + 2 * N_EDGES, rel_idx + 2 * N_EDGES,
        batch_idx + 2 * N_EDGES, dst_idx + 2 * N_EDGES,
        query_rel, rela_embed, W1, W2);

    hipMemsetAsync(out, 0, (size_t)out_size * sizeof(float), stream);
    redgnn_final_kernel<<<(N_NODES + 255) / 256, 256, 0, stream>>>(
        hB, final_b, final_e, Wc, bc, out);
}
